// ReLU2HeadAssociativeLM_34368328303122
// MI455X (gfx1250) — matrix operand order repaired and checked by reading, not compiled, not run on hardware
//
#include <hip/hip_runtime.h>

// ---------------------------------------------------------------------------
// Problem constants (from the reference)
// ---------------------------------------------------------------------------
#define VSZ 32000L
#define ESZ 512L
#define HSZ 1024L
#define MSZ 256L
#define BSZ 4L
#define TSZ 1024L

typedef __attribute__((ext_vector_type(16))) __bf16 v16bf;
typedef __attribute__((ext_vector_type(8)))  float  v8f;

union FragBF16 {
    v16bf v;
    uint4 q[2];
};

// ---------------------------------------------------------------------------
// f32 -> bf16 cast (grid-stride)
// ---------------------------------------------------------------------------
__global__ void k_cast_bf16(const float* __restrict__ src, __bf16* __restrict__ dst, long n) {
    long i = (long)blockIdx.x * blockDim.x + threadIdx.x;
    long stride = (long)gridDim.x * blockDim.x;
    for (; i < n; i += stride) dst[i] = (__bf16)src[i];
}

// ---------------------------------------------------------------------------
// Embedding gather: x[b,t,:] = emb_bf16[ids[b,t],:]
// ---------------------------------------------------------------------------
__global__ void k_embed(const int* __restrict__ ids, const __bf16* __restrict__ emb,
                        __bf16* __restrict__ x) {
    const int row = blockIdx.x;                 // 0..B*T-1
    const int id  = ids[row];
    const __bf16* src = emb + (size_t)id * ESZ;
    __bf16* dst       = x   + (size_t)row * ESZ;
    for (int j = threadIdx.x; j < ESZ; j += blockDim.x) dst[j] = src[j];
}

// ---------------------------------------------------------------------------
// Generic NT WMMA GEMM:  C[M,N] = alpha * A[M,K] * Bw[N,K]^T + bias[N]
// A, Bw row-major bf16 (K contiguous).  One wave -> 16x32 output tile
// (two v_wmma_f32_16x16x32_bf16 per K-step, A frag reused).
// Block = 256 threads = 8 waves -> 16x256 block tile.
// Fragment packing (contiguous K order, same for A and B):
//   A (16x32 bf16): lane m=lane&15; 16 contiguous K at k0+(lane>>4)*16
//   B (32x16 bf16): lane n=lane&15; 16 contiguous K at k0+(lane>>4)*16
// ---------------------------------------------------------------------------
enum { EPI_F32 = 0, EPI_BF16 = 1, EPI_RELU2_BF16 = 2 };

template <int EPI>
__global__ void __launch_bounds__(256)
k_gemm(const __bf16* __restrict__ A, const __bf16* __restrict__ Bw,
       const float* __restrict__ bias, void* __restrict__ Cv,
       int M, int N, int K, float alpha,
       long batchA, long batchB, long batchC) {
    A  += (long)blockIdx.z * batchA;
    Bw += (long)blockIdx.z * batchB;
    const int lane  = threadIdx.x & 31;
    const int wave  = threadIdx.x >> 5;
    const int m0    = blockIdx.y * 16;
    const int n0    = blockIdx.x * 256 + wave * 32;
    const int sub   = lane & 15;
    const int khalf = lane >> 4;

    const __bf16* arow  = A  + (size_t)(m0 + sub) * K;
    const __bf16* brow0 = Bw + (size_t)(n0 + sub) * K;
    const __bf16* brow1 = Bw + (size_t)(n0 + 16 + sub) * K;

    v8f acc0 = {}; v8f acc1 = {};
    for (int k0 = 0; k0 < K; k0 += 32) {
        FragBF16 fa, fb0, fb1;
        fa.q[0] = *(const uint4*)(arow + k0 + khalf * 16);
        fa.q[1] = *(const uint4*)(arow + k0 + khalf * 16 + 8);
        const __bf16* bp0 = brow0 + k0 + khalf * 16;
        const __bf16* bp1 = brow1 + k0 + khalf * 16;
        fb0.q[0] = ((const uint4*)bp0)[0]; fb0.q[1] = ((const uint4*)bp0)[1];
        fb1.q[0] = ((const uint4*)bp1)[0]; fb1.q[1] = ((const uint4*)bp1)[1];
        acc0 = __builtin_amdgcn_wmma_f32_16x16x32_bf16(false, fa.v, false, fb0.v,
                                                       (short)0, acc0, false, false);
        acc1 = __builtin_amdgcn_wmma_f32_16x16x32_bf16(false, fa.v, false, fb1.v,
                                                       (short)0, acc1, false, false);
    }

    const long cbase = (long)blockIdx.z * batchC;
    const int  colA  = n0 + sub;
    const int  colB  = colA + 16;
    const float bA = bias ? bias[colA] : 0.0f;
    const float bB = bias ? bias[colB] : 0.0f;
#pragma unroll
    for (int i = 0; i < 8; ++i) {
        const int row = m0 + i + (khalf << 3);   // lanes 16..31 hold M = i+8
        float v0 = acc0[i] * alpha + bA;
        float v1 = acc1[i] * alpha + bB;
        const long idx0 = cbase + (long)row * N + colA;
        const long idx1 = cbase + (long)row * N + colB;
        if (EPI == EPI_F32) {
            ((float*)Cv)[idx0] = v0;
            ((float*)Cv)[idx1] = v1;
        } else if (EPI == EPI_BF16) {
            ((__bf16*)Cv)[idx0] = (__bf16)v0;
            ((__bf16*)Cv)[idx1] = (__bf16)v1;
        } else {  // relu^2 -> bf16
            float r0 = fmaxf(v0, 0.0f), r1 = fmaxf(v1, 0.0f);
            ((__bf16*)Cv)[idx0] = (__bf16)(r0 * r0);
            ((__bf16*)Cv)[idx1] = (__bf16)(r1 * r1);
        }
    }
}

// ---------------------------------------------------------------------------
// Persistent GRU scan.  Single workgroup (1024 thr = 32 waves, one WGP).
// h kept in LDS as bf16 padded 16x1024 A-tile (rows 4..15 zero) + f32 copy.
// Per step: gh = h @ Whh^T via WMMA.  Loop nest is K-OUTER with 6 persistent
// accumulators per wave (48 VGPRs) so the shared A fragment lives in 8 VGPRs
// and is reused by 6 back-to-back WMMAs -- no cross-iteration hoisting, no
// scratch spills in the serial inner loop.
// ---------------------------------------------------------------------------
__global__ void __launch_bounds__(1024)
k_gru_scan(const float* __restrict__ gx,       // [B*T, 3H] (bih already added)
           const float* __restrict__ bhh,      // [3H]
           const __bf16* __restrict__ Whh,     // [3H, H] bf16
           __bf16* __restrict__ states) {      // [B*T, H] bf16
    __shared__ __align__(16) __bf16 hpad[16][HSZ];   // 32 KB
    __shared__ float ghs[BSZ][3 * HSZ];              // 48 KB
    __shared__ float hf32[BSZ][HSZ];                 // 16 KB

    const int tid  = threadIdx.x;
    const int lane = tid & 31;
    const int wave = tid >> 5;
    const int sub  = lane & 15;
    const int khalf = lane >> 4;

    for (int i = tid; i < 16 * HSZ; i += 1024) ((__bf16*)hpad)[i] = (__bf16)0.0f;
    for (int i = tid; i < BSZ * HSZ; i += 1024) ((float*)hf32)[i] = 0.0f;
    __syncthreads();

    const int nbase = wave * 96;                         // 6 tiles of 16 cols
    const __bf16* arow  = &hpad[sub][0];
    const __bf16* bbase = Whh + (size_t)(nbase + sub) * HSZ;

#pragma unroll 1
    for (int t = 0; t < TSZ; ++t) {
        // ---- phase 1: gh = hpad @ Whh^T (K-outer, 6 accumulators) ----
        v8f acc[6];
#pragma unroll
        for (int i = 0; i < 6; ++i) acc[i] = (v8f){};

#pragma unroll 1
        for (int k0 = 0; k0 < (int)HSZ; k0 += 32) {
            FragBF16 fa;
            fa.q[0] = *(const uint4*)(arow + k0 + khalf * 16);
            fa.q[1] = *(const uint4*)(arow + k0 + khalf * 16 + 8);
            const __bf16* bk = bbase + k0 + khalf * 16;
#pragma unroll
            for (int tile = 0; tile < 6; ++tile) {
                FragBF16 fb;
                fb.q[0] = ((const uint4*)(bk + (size_t)tile * 16 * HSZ))[0];
                fb.q[1] = ((const uint4*)(bk + (size_t)tile * 16 * HSZ))[1];
                acc[tile] = __builtin_amdgcn_wmma_f32_16x16x32_bf16(
                    false, fa.v, false, fb.v, (short)0, acc[tile], false, false);
            }
        }
        if (lane < 16) {           // only batch rows 0..3 are live
#pragma unroll
            for (int tile = 0; tile < 6; ++tile)
#pragma unroll
                for (int i = 0; i < 4; ++i)
                    ghs[i][nbase + tile * 16 + lane] = acc[tile][i];
        }
        __syncthreads();

        // ---- phase 2: fused GRU gate update ----
        for (int e = tid; e < (int)(BSZ * HSZ); e += 1024) {
            const int b  = e >> 10;
            const int hh = e & (HSZ - 1);
            const float* gxr = gx + (size_t)(b * TSZ + t) * (3 * HSZ);
            const float xr = gxr[hh];
            const float xz = gxr[HSZ + hh];
            const float xn = gxr[2 * HSZ + hh];
            const float hr = ghs[b][hh]            + bhh[hh];
            const float hz = ghs[b][HSZ + hh]      + bhh[HSZ + hh];
            const float hn = ghs[b][2 * HSZ + hh]  + bhh[2 * HSZ + hh];
            const float r = 1.0f / (1.0f + __expf(-(xr + hr)));
            const float z = 1.0f / (1.0f + __expf(-(xz + hz)));
            const float n = tanhf(xn + r * hn);
            const float hnew = (1.0f - z) * n + z * hf32[b][hh];
            hf32[b][hh] = hnew;
            hpad[b][hh] = (__bf16)hnew;
            states[(size_t)(b * TSZ + t) * HSZ + hh] = (__bf16)hnew;
        }
        __syncthreads();
    }
}

// ---------------------------------------------------------------------------
// gate[b,t] = sigmoid(states[b,t,:] . gate_W + gate_b)   (one wave per row)
// ---------------------------------------------------------------------------
__global__ void k_gate(const __bf16* __restrict__ states, const float* __restrict__ gW,
                       const float* __restrict__ gb, float* __restrict__ gate) {
    const int row  = blockIdx.x * 8 + (threadIdx.x >> 5);
    const int lane = threadIdx.x & 31;
    const __bf16* s = states + (size_t)row * HSZ;
    float sum = 0.0f;
    for (int j = lane; j < (int)HSZ; j += 32) sum += (float)s[j] * gW[j];
    for (int off = 16; off > 0; off >>= 1) sum += __shfl_down(sum, off, 32);
    if (lane == 0) gate[row] = 1.0f / (1.0f + __expf(-(sum + gb[0])));
}

// ---------------------------------------------------------------------------
// Masked softmax over strictly-lower-triangular scores row + gated scatter
// into logits:  out[b,q,ids[b,k]] += gate[b,q]*mem_scale * softmax(scores)[k]
// One 256-thread block per (q,b).
// ---------------------------------------------------------------------------
__global__ void k_attn(const float* __restrict__ scores, const float* __restrict__ gate,
                       const float* __restrict__ mem_scale, const int* __restrict__ ids,
                       float* __restrict__ out) {
    const int q = blockIdx.x;
    const int b = blockIdx.y;
    if (q == 0) return;                          // no valid keys -> contribution 0
    const float* srow = scores + ((size_t)(b * TSZ + q)) * TSZ;
    __shared__ float red[256];
    const int tid = threadIdx.x;

    float m = -3.0e38f;
    for (int k = tid; k < q; k += 256) m = fmaxf(m, srow[k]);
    red[tid] = m; __syncthreads();
    for (int s = 128; s > 0; s >>= 1) {
        if (tid < s) red[tid] = fmaxf(red[tid], red[tid + s]);
        __syncthreads();
    }
    m = red[0]; __syncthreads();

    float sum = 0.0f;
    for (int k = tid; k < q; k += 256) sum += __expf(srow[k] - m);
    red[tid] = sum; __syncthreads();
    for (int s = 128; s > 0; s >>= 1) {
        if (tid < s) red[tid] += red[tid + s];
        __syncthreads();
    }
    sum = red[0];

    const float coef = gate[b * TSZ + q] * mem_scale[0] / fmaxf(sum, 1e-6f);
    const int* idrow = ids + b * TSZ;
    float* orow = out + ((size_t)(b * TSZ + q)) * VSZ;
    for (int k = tid; k < q; k += 256) {
        const float ga = __expf(srow[k] - m) * coef;
        atomicAdd(&orow[idrow[k]], ga);
    }
}

// ---------------------------------------------------------------------------
// Host launcher
// ---------------------------------------------------------------------------
extern "C" void kernel_launch(void* const* d_in, const int* in_sizes, int n_in,
                              void* d_out, int out_size, void* d_ws, size_t ws_size,
                              hipStream_t stream) {
    (void)in_sizes; (void)n_in; (void)out_size; (void)ws_size;
    const int*   ids      = (const int*)d_in[0];
    const float* emb_W    = (const float*)d_in[1];
    const float* Wih      = (const float*)d_in[2];
    const float* Whh      = (const float*)d_in[3];
    const float* bih      = (const float*)d_in[4];
    const float* bhh      = (const float*)d_in[5];
    const float* qW       = (const float*)d_in[6];
    const float* qb       = (const float*)d_in[7];
    const float* kW       = (const float*)d_in[8];
    const float* kb       = (const float*)d_in[9];
    const float* gateW    = (const float*)d_in[10];
    const float* gateb    = (const float*)d_in[11];
    const float* hfcW     = (const float*)d_in[12];
    const float* hfcb     = (const float*)d_in[13];
    const float* hpW      = (const float*)d_in[14];
    const float* hpb      = (const float*)d_in[15];
    const float* out_bias = (const float*)d_in[16];
    const float* memsc    = (const float*)d_in[17];
    float* out = (float*)d_out;

    // ---- workspace bump allocator (256B aligned) ----
    char* ws = (char*)d_ws;
    size_t off = 0;
    auto alloc = [&](size_t bytes) -> char* {
        char* p = ws + off;
        off += (bytes + 255) & ~(size_t)255;
        return p;
    };
    __bf16* emb_bf  = (__bf16*)alloc(VSZ * ESZ * 2);
    __bf16* Wih_bf  = (__bf16*)alloc(3 * HSZ * ESZ * 2);
    __bf16* Whh_bf  = (__bf16*)alloc(3 * HSZ * HSZ * 2);
    __bf16* hfcW_bf = (__bf16*)alloc(4 * ESZ * HSZ * 2);
    __bf16* hpW_bf  = (__bf16*)alloc(ESZ * 4 * ESZ * 2);
    __bf16* qW_bf   = (__bf16*)alloc(MSZ * HSZ * 2);
    __bf16* kW_bf   = (__bf16*)alloc(MSZ * HSZ * 2);
    __bf16* x_bf    = (__bf16*)alloc(BSZ * TSZ * ESZ * 2);
    __bf16* st_bf   = (__bf16*)alloc(BSZ * TSZ * HSZ * 2);
    float*  gatev   = (float*)alloc(BSZ * TSZ * 4);
    // BIG region: gx (f32, 50.3MB) during the scan; recycled afterwards
    char*   big     = alloc(BSZ * TSZ * 3 * HSZ * 4);
    float*  gx      = (float*)big;
    float*  scoresv = (float*)big;                                   // 16.8 MB
    __bf16* hf_bf   = (__bf16*)(big + (size_t)BSZ * TSZ * TSZ * 4);  // 16.8 MB
    __bf16* base_bf = (__bf16*)(big + 33554432);                     //  4.2 MB
    __bf16* q_bf    = (__bf16*)(big + 37748736);                     //  2.1 MB
    __bf16* k_bf    = (__bf16*)(big + 39845888);                     //  2.1 MB

    // ---- weight / activation casts to bf16 ----
    k_cast_bf16<<<4096, 256, 0, stream>>>(emb_W, emb_bf, VSZ * ESZ);
    k_cast_bf16<<<2048, 256, 0, stream>>>(Wih, Wih_bf, 3 * HSZ * ESZ);
    k_cast_bf16<<<2048, 256, 0, stream>>>(Whh, Whh_bf, 3 * HSZ * HSZ);
    k_cast_bf16<<<2048, 256, 0, stream>>>(hfcW, hfcW_bf, 4 * ESZ * HSZ);
    k_cast_bf16<<<2048, 256, 0, stream>>>(hpW, hpW_bf, ESZ * 4 * ESZ);
    k_cast_bf16<<<512, 256, 0, stream>>>(qW, qW_bf, MSZ * HSZ);
    k_cast_bf16<<<512, 256, 0, stream>>>(kW, kW_bf, MSZ * HSZ);

    // ---- embedding gather ----
    k_embed<<<BSZ * TSZ, 256, 0, stream>>>(ids, emb_bf, x_bf);

    // ---- gx = x @ Wih^T + bih  (hoisted out of the serial scan) ----
    k_gemm<EPI_F32><<<dim3(3 * HSZ / 256, BSZ * TSZ / 16, 1), 256, 0, stream>>>(
        x_bf, Wih_bf, bih, gx, BSZ * TSZ, 3 * HSZ, ESZ, 1.0f, 0, 0, 0);

    // ---- serial GRU scan (persistent single-WGP WMMA kernel) ----
    k_gru_scan<<<1, 1024, 0, stream>>>(gx, bhh, Whh_bf, st_bf);

    // ---- hf = relu(states @ hfc_W^T + hfc_b)^2 ----
    k_gemm<EPI_RELU2_BF16><<<dim3(4 * ESZ / 256, BSZ * TSZ / 16, 1), 256, 0, stream>>>(
        st_bf, hfcW_bf, hfcb, hf_bf, BSZ * TSZ, 4 * ESZ, HSZ, 1.0f, 0, 0, 0);

    // ---- base = hf @ hp_W^T + hp_b ----
    k_gemm<EPI_BF16><<<dim3(ESZ / 256, BSZ * TSZ / 16, 1), 256, 0, stream>>>(
        hf_bf, hpW_bf, hpb, base_bf, BSZ * TSZ, ESZ, 4 * ESZ, 1.0f, 0, 0, 0);

    // ---- logits = base @ emb^T + out_bias  (dominant GEMM, 134 GF) ----
    k_gemm<EPI_F32><<<dim3(VSZ / 256, BSZ * TSZ / 16, 1), 256, 0, stream>>>(
        base_bf, emb_bf, out_bias, out, BSZ * TSZ, VSZ, ESZ, 1.0f, 0, 0, 0);

    // ---- q / k projections ----
    k_gemm<EPI_BF16><<<dim3(MSZ / 256, BSZ * TSZ / 16, 1), 256, 0, stream>>>(
        st_bf, qW_bf, qb, q_bf, BSZ * TSZ, MSZ, HSZ, 1.0f, 0, 0, 0);
    k_gemm<EPI_BF16><<<dim3(MSZ / 256, BSZ * TSZ / 16, 1), 256, 0, stream>>>(
        st_bf, kW_bf, kb, k_bf, BSZ * TSZ, MSZ, HSZ, 1.0f, 0, 0, 0);

    // ---- scores[b] = q[b] @ k[b]^T / sqrt(M)  (batched via blockIdx.z) ----
    k_gemm<EPI_F32><<<dim3(TSZ / 256, TSZ / 16, BSZ), 256, 0, stream>>>(
        q_bf, k_bf, nullptr, scoresv, TSZ, TSZ, MSZ, 0.0625f,
        TSZ * MSZ, TSZ * MSZ, TSZ * TSZ);

    // ---- gate vector ----
    k_gate<<<BSZ * TSZ / 8, 256, 0, stream>>>(st_bf, gateW, gateb, gatev);

    // ---- masked softmax + gated scatter-add into logits ----
    k_attn<<<dim3(TSZ, BSZ), 256, 0, stream>>>(scoresv, gatev, memsc, ids, out);
}
